// KGAT_19731079758337
// MI455X (gfx1250) — matrix operand order repaired and checked by reading, not compiled, not run on hardware
//
#include <hip/hip_runtime.h>
#include <math.h>

#define NEG_SLOPE 0.01f

typedef __attribute__((ext_vector_type(16))) _Float16 v16h;
typedef __attribute__((ext_vector_type(8)))  float    v8f;

// ---------------- zero workspace (side accumulator) ----------------
__global__ void kgat_zero_kernel(float4* __restrict__ p, int n4) {
    int i = blockIdx.x * blockDim.x + threadIdx.x;
    if (i < n4) p[i] = make_float4(0.f, 0.f, 0.f, 0.f);
}

// ---------------- copy layer-0 embeddings into out[:, 0:64] ----------------
// out row stride = 256 floats = 64 float4
__global__ void kgat_copy_l0_kernel(const float* __restrict__ ego,
                                    float* __restrict__ out, int n) {
    int i = blockIdx.x * blockDim.x + threadIdx.x;  // one float4 slot
    int node = i >> 4, c4 = i & 15;
    if (node < n) {
        float4 v = ((const float4*)ego)[node * 16 + c4];
        ((float4*)out)[node * 64 + c4] = v;
    }
}

// ---------------- convert 3 layers of W1/W2 to f16 once ----------------
__global__ void kgat_wcvt_kernel(const float* __restrict__ W1,
                                 const float* __restrict__ W2,
                                 _Float16* __restrict__ w1h,
                                 _Float16* __restrict__ w2h, int tot) {
    int i = blockIdx.x * blockDim.x + threadIdx.x;
    if (i < tot) {
        w1h[i] = (_Float16)W1[i];
        w2h[i] = (_Float16)W2[i];
    }
}

// ---------------- SpMM: side[r] += val * ego[c] ----------------
// Wave handles 32 edges: coalesced metadata load, readlane broadcast
// (scalar row/col -> SGPR addressing), per-edge 256B row prefetch,
// float2 gather + 2 native f32 atomics per lane (L2-resident target).
__global__ __launch_bounds__(256) void kgat_spmm_kernel(
    const int* __restrict__ rows, const int* __restrict__ cols,
    const float* __restrict__ vals, const float* __restrict__ x,
    float* __restrict__ side, int nE) {
    const int lane = threadIdx.x & 31;
    const int wave = (blockIdx.x * blockDim.x + threadIdx.x) >> 5;
    const int e0 = wave * 32;

    // each lane owns one edge's metadata (coalesced)
    int e = e0 + lane;
    int r = 0, c = 0;
    float v = 0.f;
    if (e < nE) {
        r = rows[e];
        c = cols[e];
        v = vals[e];
        // prefetch this edge's gather row (2 x 128B lines) ahead of use
        __builtin_prefetch(x + (size_t)c * 64, 0, 3);
        __builtin_prefetch(x + (size_t)c * 64 + 32, 0, 3);
    }

#pragma unroll
    for (int j = 0; j < 32; ++j) {
        if (e0 + j >= nE) break;                    // wave-uniform exit
        int rj = __builtin_amdgcn_readlane(r, j);   // scalar row
        int cj = __builtin_amdgcn_readlane(c, j);   // scalar col
        float vj = __int_as_float(__builtin_amdgcn_readlane(__float_as_int(v), j));
        float2 xv = ((const float2*)x)[(size_t)cj * 32 + lane];
        float* dst = side + (size_t)rj * 64 + lane * 2;
        __hip_atomic_fetch_add(dst,     vj * xv.x, __ATOMIC_RELAXED, __HIP_MEMORY_SCOPE_AGENT);
        __hip_atomic_fetch_add(dst + 1, vj * xv.y, __ATOMIC_RELAXED, __HIP_MEMORY_SCOPE_AGENT);
    }
}

// ---------------- fused dense layer: 2x GEMM (WMMA f16) + bias + lrelu +
//                  combine + row-normalize; 8 waves x 16 rows per block ------
__global__ __launch_bounds__(256) void kgat_fused_layer_kernel(
    const float* __restrict__ ego, const float* __restrict__ side,
    const _Float16* __restrict__ w1h, const float* __restrict__ b1k,
    const _Float16* __restrict__ w2h, const float* __restrict__ b2k,
    float* __restrict__ egoNext, float* __restrict__ outCol, int n) {
    __shared__ __align__(16) _Float16 sW1[64 * 64];
    __shared__ __align__(16) _Float16 sW2[64 * 64];
    __shared__ float sB1[64];
    __shared__ float sB2[64];
    __shared__ __align__(16) _Float16 sAdd[8][16 * 64];
    __shared__ __align__(16) _Float16 sProd[8][16 * 64];
    __shared__ __align__(16) float sEgo[8][16 * 64];
    __shared__ float sInv[8][16];

    const int tid  = threadIdx.x;
    const int w    = tid >> 5;
    const int lane = tid & 31;
    const int half = lane >> 4;   // lane half, per ISA A/B fragment layout
    const int l16  = lane & 15;

    // stage pre-converted f16 weights (uint4 moves) + biases into LDS
    {
        const uint4* g1 = (const uint4*)w1h;
        const uint4* g2 = (const uint4*)w2h;
        uint4* s1 = (uint4*)sW1;
        uint4* s2 = (uint4*)sW2;
#pragma unroll
        for (int i = tid; i < 512; i += 256) {   // 4096 f16 = 512 uint4
            s1[i] = g1[i];
            s2[i] = g2[i];
        }
    }
    if (tid < 64) { sB1[tid] = b1k[tid]; sB2[tid] = b2k[tid]; }

    const int rowBase = blockIdx.x * 128 + w * 16;

    // load 16x64 ego/side rows, build add = ego+side and prod = ego*side (f16)
#pragma unroll
    for (int j = 0; j < 8; ++j) {
        int idx = lane + 32 * j;          // 256 float4 slots = 16 rows x 16
        int r = idx >> 4, c4 = idx & 15;
        int node = rowBase + r;
        if (node > n - 1) node = n - 1;   // clamp (stores are guarded later)
        float4 e = ((const float4*)ego)[node * 16 + c4];
        float4 s = ((const float4*)side)[node * 16 + c4];
        int base = r * 64 + c4 * 4;
        sAdd[w][base + 0] = (_Float16)(e.x + s.x);
        sAdd[w][base + 1] = (_Float16)(e.y + s.y);
        sAdd[w][base + 2] = (_Float16)(e.z + s.z);
        sAdd[w][base + 3] = (_Float16)(e.w + s.w);
        sProd[w][base + 0] = (_Float16)(e.x * s.x);
        sProd[w][base + 1] = (_Float16)(e.y * s.y);
        sProd[w][base + 2] = (_Float16)(e.z * s.z);
        sProd[w][base + 3] = (_Float16)(e.w * s.w);
    }
    __syncthreads();

    v8f accS[4], accB[4];
    {
        v8f z = {};
#pragma unroll
        for (int t = 0; t < 4; ++t) { accS[t] = z; accB[t] = z; }
    }

#pragma unroll
    for (int ks = 0; ks < 2; ++ks) {       // K = 64 -> two K=32 WMMA steps
        const int kbase = ks * 32;
        // A fragments, 16-bit 16x32 layout:
        //  lanes 0-15: VGPR0-3 -> K=0..7, VGPR4-7 -> K=16..23
        //  lanes16-31: VGPR0-3 -> K=8..15, VGPR4-7 -> K=24..31
        v16h aAdd, aProd;
#pragma unroll
        for (int i = 0; i < 8; ++i) {
            int kk = kbase + ((i < 4) ? (2 * i + 8 * half)
                                      : (16 + 2 * (i - 4) + 8 * half));
            aAdd[2 * i]      = sAdd[w][l16 * 64 + kk];
            aAdd[2 * i + 1]  = sAdd[w][l16 * 64 + kk + 1];
            aProd[2 * i]     = sProd[w][l16 * 64 + kk];
            aProd[2 * i + 1] = sProd[w][l16 * 64 + kk + 1];
        }
#pragma unroll
        for (int tn = 0; tn < 4; ++tn) {   // 64 output cols = 4 N-tiles
            const int nn = tn * 16 + l16;
            // B = W^T (out = x @ W.T): B[k][n] = W[n][k]; k-pairs contiguous
            // in row-major W. lanes 0-15: K=0..7,16..23, lanes 16-31: K=8..15,24..31.
            v16h bS, bB;
#pragma unroll
            for (int i = 0; i < 8; ++i) {
                int kk = kbase + ((i < 4) ? (2 * i + 8 * half)
                                          : (16 + 2 * (i - 4) + 8 * half));
                bS[2 * i]     = sW1[nn * 64 + kk];
                bS[2 * i + 1] = sW1[nn * 64 + kk + 1];
                bB[2 * i]     = sW2[nn * 64 + kk];
                bB[2 * i + 1] = sW2[nn * 64 + kk + 1];
            }
            accS[tn] = __builtin_amdgcn_wmma_f32_16x16x32_f16(
                false, aAdd, false, bS, (short)0, accS[tn], false, false);
            accB[tn] = __builtin_amdgcn_wmma_f32_16x16x32_f16(
                false, aProd, false, bB, (short)0, accB[tn], false, false);
        }
    }

    // epilogue: bias + leaky-relu both paths, combine, stash f32 tile in LDS
    // C/D layout: VGPR i -> row m = i + 8*half, col = l16 (within tile tn)
#pragma unroll
    for (int tn = 0; tn < 4; ++tn) {
        const int nn = tn * 16 + l16;
        const float bb1 = sB1[nn], bb2 = sB2[nn];
#pragma unroll
        for (int i = 0; i < 8; ++i) {
            float s = accS[tn][i] + bb1;
            s = (s > 0.f) ? s : NEG_SLOPE * s;
            float b = accB[tn][i] + bb2;
            b = (b > 0.f) ? b : NEG_SLOPE * b;
            sEgo[w][(i + 8 * half) * 64 + nn] = s + b;
        }
    }
    __syncthreads();

    // row L2 norms: lane l16 owns row l16; halves split the 64 cols, combine
    float ss = 0.f;
    {
        const float* rowp = &sEgo[w][l16 * 64 + half * 32];
#pragma unroll
        for (int c = 0; c < 32; c += 4) {
            float4 vv = *(const float4*)(rowp + c);
            ss += vv.x * vv.x + vv.y * vv.y + vv.z * vv.z + vv.w * vv.w;
        }
    }
    ss += __shfl_xor(ss, 16, 32);
    if (lane < 16) sInv[w][lane] = 1.0f / fmaxf(sqrtf(ss), 1e-12f);
    __syncthreads();

    // write next-layer ego (unnormalized) + normalized output slice
#pragma unroll
    for (int j = 0; j < 8; ++j) {
        int idx = lane + 32 * j;
        int r = idx >> 4, c4 = idx & 15;
        int node = rowBase + r;
        if (node < n) {
            float4 vv = ((const float4*)&sEgo[w][0])[r * 16 + c4];
            ((float4*)egoNext)[node * 16 + c4] = vv;
            float iv = sInv[w][r];
            float4 nv = make_float4(vv.x * iv, vv.y * iv, vv.z * iv, vv.w * iv);
            ((float4*)outCol)[node * 64 + c4] = nv;   // row stride 256 floats
        }
    }
}

extern "C" void kernel_launch(void* const* d_in, const int* in_sizes, int n_in,
                              void* d_out, int out_size, void* d_ws, size_t ws_size,
                              hipStream_t stream) {
    const int*   edge_rows = (const int*)d_in[0];
    const int*   edge_cols = (const int*)d_in[1];
    const float* edge_vals = (const float*)d_in[2];
    const float* ego0      = (const float*)d_in[3];
    const float* W1        = (const float*)d_in[4];   // [3][64][64]
    const float* b1        = (const float*)d_in[5];   // [3][64]
    const float* W2        = (const float*)d_in[6];
    const float* b2        = (const float*)d_in[7];
    float* out = (float*)d_out;

    const int nE = in_sizes[0];
    const int n  = in_sizes[3] / 64;                  // 150000 nodes

    float* side = (float*)d_ws;                       // n*64 f32
    float* egoA = side + (size_t)n * 64;              // n*64 f32
    float* egoB = egoA + (size_t)n * 64;              // n*64 f32
    _Float16* w1h = (_Float16*)(egoB + (size_t)n * 64);  // 3*64*64 f16
    _Float16* w2h = w1h + 3 * 64 * 64;                   // 3*64*64 f16

    // out[:, 0:64] = ego0
    {
        int tot = n * 16;
        kgat_copy_l0_kernel<<<(tot + 255) / 256, 256, 0, stream>>>(ego0, out, n);
    }
    // pre-convert all weights to f16 once
    {
        int tot = 3 * 64 * 64;
        kgat_wcvt_kernel<<<(tot + 255) / 256, 256, 0, stream>>>(W1, W2, w1h, w2h, tot);
    }

    const int nzero4      = n * 16;
    const int waves       = (nE + 31) / 32;
    const int spmmBlocks  = (waves + 7) / 8;          // 8 waves per 256-thr block
    const int fusedBlocks = (n + 127) / 128;          // 128 rows per block

    const float* egoCur = ego0;
    float* ping = egoA;
    float* pong = egoB;
    for (int k = 0; k < 3; ++k) {
        kgat_zero_kernel<<<(nzero4 + 255) / 256, 256, 0, stream>>>(
            (float4*)side, nzero4);
        kgat_spmm_kernel<<<spmmBlocks, 256, 0, stream>>>(
            edge_rows, edge_cols, edge_vals, egoCur, side, nE);
        kgat_fused_layer_kernel<<<fusedBlocks, 256, 0, stream>>>(
            egoCur, side, w1h + k * 4096, b1 + k * 64, w2h + k * 4096, b2 + k * 64,
            ping, out + 64 * (k + 1), n);
        egoCur = ping;
        float* t = ping; ping = pong; pong = t;
    }
}
